// MAMLAwareGANLoss_14577119003481
// MI455X (gfx1250) — matrix operand order repaired and checked by reading, not compiled, not run on hardware
//
#include <hip/hip_runtime.h>
#include <hip/hip_bf16.h>

// ---------------------------------------------------------------------------
// MAML-aware GAN loss, MI455X (gfx1250, wave32).
// fake = tanh(noise[1024x256] @ Wg[256x16384])  -- bf16 hi/lo split WMMA GEMM
// d_fake fused into GEMM epilogue; maze penalty via bitwise flood fill.
// ---------------------------------------------------------------------------

typedef __bf16 bf16_t;
typedef bf16_t v16bf __attribute__((ext_vector_type(16)));
typedef float  v8f   __attribute__((ext_vector_type(8)));

#define BATCH 1024
#define ZDIM  256
#define NDIM  16384
#define MBLK  128
#define NBLK  128
#define KSTEP 32
#define LDT   40   // padded k-stride (ushorts) to dodge LDS bank conflicts

__device__ __forceinline__ unsigned short bf_hi(float x) {
  union { float f; unsigned u; } c; c.f = x;
  return (unsigned short)(c.u >> 16);
}
__device__ __forceinline__ float bf_f(unsigned short h) {
  union { unsigned u; float f; } c; c.u = ((unsigned)h) << 16;
  return c.f;
}

union FragU { uint4 q[2]; v16bf v; };

// ---------------------------------------------------------------------------
// Kernel 0: zero the small accumulators in workspace.
// ---------------------------------------------------------------------------
__global__ void zero_kernel(float* p, int n) {
  int i = blockIdx.x * blockDim.x + threadIdx.x;
  if (i < n) p[i] = 0.0f;
}

// ---------------------------------------------------------------------------
// Kernel 1: fake = tanh(noise @ Wg), fused d_fake = fake @ Wd (atomics).
// Grid: (NDIM/128, BATCH/128), 256 threads (8 waves, 2x4 wave grid).
// Each wave: 64x32 C tile = 4x2 fragments of 16x16, K accumulated via
// 3x V_WMMA_F32_16X16X32_BF16 per fragment per k-step (hi/lo f32 split).
// ---------------------------------------------------------------------------
__global__ __launch_bounds__(256) void gen_gemm_kernel(
    const float* __restrict__ noise, const float* __restrict__ Wg,
    const float* __restrict__ Wd, float* __restrict__ fake,
    float* __restrict__ dfake)
{
  __shared__ __align__(16) unsigned short sAh[MBLK * LDT]; // A hi  [m][k]
  __shared__ __align__(16) unsigned short sAl[MBLK * LDT]; // A lo  [m][k]
  __shared__ __align__(16) unsigned short sBh[NBLK * LDT]; // B hi  [n][k] (transposed)
  __shared__ __align__(16) unsigned short sBl[NBLK * LDT]; // B lo  [n][k]
  __shared__ float sRow[MBLK];

  const int tid   = threadIdx.x;
  const int wave  = tid >> 5;
  const int lane  = tid & 31;
  const int l16   = lane & 15;
  const int half  = lane >> 4;
  const int waveM = wave >> 2;  // 0..1
  const int waveN = wave & 3;   // 0..3
  const int nBlock = blockIdx.x * NBLK;
  const int mBlock = blockIdx.y * MBLK;

  if (tid < MBLK) sRow[tid] = 0.0f;

  v8f acc[4][2];
#pragma unroll
  for (int mf = 0; mf < 4; ++mf)
#pragma unroll
    for (int nf = 0; nf < 2; ++nf)
#pragma unroll
      for (int v = 0; v < 8; ++v) acc[mf][nf][v] = 0.0f;

  for (int kBase = 0; kBase < ZDIM; kBase += KSTEP) {
    __syncthreads();
    // Stage A tile (128 x 32 f32) -> hi/lo bf16, row-major [m][k]
#pragma unroll
    for (int i = 0; i < 16; ++i) {
      int idx = tid + i * 256;
      int r = idx >> 5, c = idx & 31;
      float x = noise[(mBlock + r) * ZDIM + kBase + c];
      unsigned short h = bf_hi(x);
      sAh[r * LDT + c] = h;
      sAl[r * LDT + c] = bf_hi(x - bf_f(h));
    }
    // Stage B tile (32 x 128 f32) -> hi/lo bf16, transposed [n][k]
#pragma unroll
    for (int i = 0; i < 16; ++i) {
      int idx = tid + i * 256;
      int k = idx >> 7, n = idx & 127;
      float x = Wg[(kBase + k) * NDIM + nBlock + n];
      unsigned short h = bf_hi(x);
      sBh[n * LDT + k] = h;
      sBl[n * LDT + k] = bf_hi(x - bf_f(h));
    }
    __syncthreads();

    // B fragments: lane (n = l16, half), VGPRs 0-3 -> K half*8..+7, VGPRs 4-7 -> K 16+half*8..+7
    FragU bh[2], bl[2];
#pragma unroll
    for (int nf = 0; nf < 2; ++nf) {
      int n = waveN * 32 + nf * 16 + l16;
      const uint4* ph = (const uint4*)(sBh + n * LDT + half * 8);
      bh[nf].q[0] = ph[0]; bh[nf].q[1] = *(const uint4*)(sBh + n * LDT + 16 + half * 8);
      const uint4* pl = (const uint4*)(sBl + n * LDT + half * 8);
      bl[nf].q[0] = pl[0]; bl[nf].q[1] = *(const uint4*)(sBl + n * LDT + 16 + half * 8);
    }
    // A fragments: lane (m = l16, half), VGPRs 0-3 -> K half*8..+7,
    // VGPRs 4-7 -> K 16+half*8..+7 (two 16B chunks).
#pragma unroll
    for (int mf = 0; mf < 4; ++mf) {
      int m = waveM * 64 + mf * 16 + l16;
      FragU ah, al;
      ah.q[0] = *(const uint4*)(sAh + m * LDT + half * 8);
      ah.q[1] = *(const uint4*)(sAh + m * LDT + 16 + half * 8);
      al.q[0] = *(const uint4*)(sAl + m * LDT + half * 8);
      al.q[1] = *(const uint4*)(sAl + m * LDT + 16 + half * 8);
#pragma unroll
      for (int nf = 0; nf < 2; ++nf) {
        acc[mf][nf] = __builtin_amdgcn_wmma_f32_16x16x32_bf16(
            false, ah.v, false, bh[nf].v, (short)0, acc[mf][nf], false, false);
        acc[mf][nf] = __builtin_amdgcn_wmma_f32_16x16x32_bf16(
            false, al.v, false, bh[nf].v, (short)0, acc[mf][nf], false, false);
        acc[mf][nf] = __builtin_amdgcn_wmma_f32_16x16x32_bf16(
            false, ah.v, false, bl[nf].v, (short)0, acc[mf][nf], false, false);
      }
    }
  }

  // Epilogue: tanh, store fake, fused d_fake row reduction.
  // C layout: lane l16 = column, VGPR v -> row v + half*8.
  float dsum[4][8];
#pragma unroll
  for (int mf = 0; mf < 4; ++mf)
#pragma unroll
    for (int v = 0; v < 8; ++v) dsum[mf][v] = 0.0f;

#pragma unroll
  for (int nf = 0; nf < 2; ++nf) {
    int n = nBlock + waveN * 32 + nf * 16 + l16;
    float wd = Wd[n];
#pragma unroll
    for (int mf = 0; mf < 4; ++mf) {
      int m0 = mBlock + waveM * 64 + mf * 16 + half * 8;
#pragma unroll
      for (int v = 0; v < 8; ++v) {
        float f = tanhf(acc[mf][nf][v]);
        fake[(m0 + v) * NDIM + n] = f;
        dsum[mf][v] += f * wd;
      }
    }
  }
#pragma unroll
  for (int mf = 0; mf < 4; ++mf)
#pragma unroll
    for (int v = 0; v < 8; ++v) {
      float sv = dsum[mf][v];
      sv += __shfl_xor(sv, 1);
      sv += __shfl_xor(sv, 2);
      sv += __shfl_xor(sv, 4);
      sv += __shfl_xor(sv, 8);
      if (l16 == 0)
        atomicAdd(&sRow[waveM * 64 + mf * 16 + half * 8 + v], sv);
    }
  __syncthreads();
  if (tid < MBLK) atomicAdd(&dfake[mBlock + tid], sRow[tid]);
}

// ---------------------------------------------------------------------------
// Kernel 2: per-maze penalty. Grid: BATCH blocks of 128 threads (thread = row).
// Bitmask flood fill in LDS; early exit when a maze has zero walls (penalty
// is then provably 0: walls_around==0 and density==0).
// ---------------------------------------------------------------------------
__global__ __launch_bounds__(128) void maze_penalty_kernel(
    const float* __restrict__ fake, float* __restrict__ solv_sum,
    float* __restrict__ wall_total)
{
  __shared__ unsigned int vis[128][4];
  __shared__ unsigned int opn[128][4];
  __shared__ int s_changed, s_narrow, s_wall;

  const int r    = threadIdx.x;
  const int wv   = r >> 5;
  const int lane = r & 31;
  const int maze = blockIdx.x;
  const float* mz = fake + (size_t)maze * NDIM;

  if (r == 0) { s_narrow = 0; s_wall = 0; }
  // Build open-cell bitmasks with coalesced loads + wave ballots:
  // thread r owns column r; wave wv supplies 32-bit word wv of each row.
  for (int i = 0; i < 128; ++i) {
    float val = mz[i * 128 + r];
    unsigned long long b = __ballot(val != 1.0f);
    if (lane == 0) opn[i][wv] = (unsigned)b;
  }
#pragma unroll
  for (int w = 0; w < 4; ++w) vis[r][w] = 0u;
  if (r == 1) vis[1][0] = 2u;  // start at (1,1)
  __syncthreads();

  unsigned o[4];
  int wallc = 0;
#pragma unroll
  for (int w = 0; w < 4; ++w) { o[w] = opn[r][w]; wallc += __popc(~o[w]); }
  atomicAdd(&s_wall, wallc);
  __syncthreads();

  if (s_wall != 0) {
    // Flood fill: visited' = visited | (dilate(visited) & open)
    for (;;) {
      if (r == 0) s_changed = 0;
      __syncthreads();
      unsigned v0[4], vm[4], vp[4], nv[4];
      int ch = 0;
#pragma unroll
      for (int w = 0; w < 4; ++w) {
        v0[w] = vis[r][w];
        vm[w] = (r > 0)   ? vis[r - 1][w] : 0u;
        vp[w] = (r < 127) ? vis[r + 1][w] : 0u;
      }
#pragma unroll
      for (int w = 0; w < 4; ++w) {
        unsigned lf = (v0[w] << 1) | (w > 0 ? (v0[w - 1] >> 31) : 0u);
        unsigned rt = (v0[w] >> 1) | (w < 3 ? (v0[w + 1] << 31) : 0u);
        unsigned dil = v0[w] | lf | rt | vm[w] | vp[w];
        nv[w] = v0[w] | (dil & o[w]);
        ch |= (nv[w] != v0[w]);
      }
      __syncthreads();
#pragma unroll
      for (int w = 0; w < 4; ++w) vis[r][w] = nv[w];
      if (ch) atomicOr(&s_changed, 1);
      __syncthreads();
      int done = (s_changed == 0);
      __syncthreads();
      if (done) break;
    }
    // narrow = count(visited & (>=3 wall neighbors))
    unsigned wl[4], up[4], dn[4];
#pragma unroll
    for (int w = 0; w < 4; ++w) {
      wl[w] = ~o[w];
      up[w] = (r > 0)   ? ~opn[r - 1][w] : 0u;
      dn[w] = (r < 127) ? ~opn[r + 1][w] : 0u;
    }
    int narrow = 0;
#pragma unroll
    for (int w = 0; w < 4; ++w) {
      unsigned lf = (wl[w] << 1) | (w > 0 ? (wl[w - 1] >> 31) : 0u);
      unsigned rt = (wl[w] >> 1) | (w < 3 ? (wl[w + 1] << 31) : 0u);
      unsigned a3 = (up[w] & dn[w] & lf) | (up[w] & dn[w] & rt) |
                    (up[w] & lf & rt)    | (dn[w] & lf & rt);
      narrow += __popc(vis[r][w] & a3);
    }
    atomicAdd(&s_narrow, narrow);
    __syncthreads();
    if (r == 0) {
      float pen = (((float)s_wall * (1.0f / 16384.0f)) > 0.5f ? 1.0f : 0.0f)
                  + 0.1f * (float)s_narrow;
      atomicAdd(solv_sum, pen);
      atomicAdd(wall_total, (float)s_wall);
    }
  }
  // s_wall == 0 -> penalty exactly 0, nothing to accumulate.
}

// ---------------------------------------------------------------------------
// Kernel 3: g_loss reduction + final combine.
// ---------------------------------------------------------------------------
__global__ __launch_bounds__(256) void finalize_kernel(
    const float* __restrict__ dfake, const float* __restrict__ solv_sum,
    const float* __restrict__ wall_total, const float* __restrict__ maml,
    const float* __restrict__ cdiff, float* __restrict__ out)
{
  __shared__ float red[256];
  int tid = threadIdx.x;
  float s = 0.0f;
  for (int i = tid; i < BATCH; i += 256) {
    float t = -dfake[i];  // softplus(-d_fake), numerically stable
    s += fmaxf(t, 0.0f) + log1pf(expf(-fabsf(t)));
  }
  red[tid] = s;
  __syncthreads();
  for (int st = 128; st > 0; st >>= 1) {
    if (tid < st) red[tid] += red[tid + st];
    __syncthreads();
  }
  if (tid == 0) {
    float g    = red[0] / (float)BATCH;
    float solv = solv_sum[0] / (float)BATCH;
    float cur  = wall_total[0] / ((float)BATCH * (float)NDIM);
    float d    = cur - cdiff[0];
    float p    = maml[0];
    float w_s  = (p < 0.4f) ? 0.8f : ((p > 0.6f) ? 0.4f : 0.6f);
    float w_d  = (p < 0.4f) ? 0.05f : ((p > 0.6f) ? 0.2f : 0.1f);
    out[0] = g + w_s * solv + w_d * d * d;
  }
}

// ---------------------------------------------------------------------------
extern "C" void kernel_launch(void* const* d_in, const int* in_sizes, int n_in,
                              void* d_out, int out_size, void* d_ws, size_t ws_size,
                              hipStream_t stream) {
  (void)in_sizes; (void)n_in; (void)out_size; (void)ws_size;
  const float* noise = (const float*)d_in[0];
  // d_in[1] = real_mazes: contributes 0.0 * sum(d_real) -> skipped (finite inputs)
  const float* Wg    = (const float*)d_in[2];
  const float* Wd    = (const float*)d_in[3];
  const float* maml  = (const float*)d_in[4];
  const float* cdiff = (const float*)d_in[5];

  float* fake  = (float*)d_ws;                         // 1024*16384 f32 (64 MB)
  float* dfake = fake + (size_t)BATCH * NDIM;          // 1024 f32
  float* solv  = dfake + BATCH;                        // 1 f32
  float* wallt = solv + 1;                             // 1 f32

  zero_kernel<<<(BATCH + 2 + 255) / 256, 256, 0, stream>>>(dfake, BATCH + 2);
  gen_gemm_kernel<<<dim3(NDIM / NBLK, BATCH / MBLK), 256, 0, stream>>>(
      noise, Wg, Wd, fake, dfake);
  maze_penalty_kernel<<<BATCH, 128, 0, stream>>>(fake, solv, wallt);
  finalize_kernel<<<1, 256, 0, stream>>>(dfake, solv, wallt, maml, cdiff,
                                         (float*)d_out);
}
